// T5Attention_69758858821737
// MI455X (gfx1250) — matrix operand order repaired and checked by reading, not compiled, not run on hardware
//
#include <hip/hip_runtime.h>

typedef __attribute__((ext_vector_type(16))) __bf16 v16bf;
typedef __attribute__((ext_vector_type(8)))  __bf16 v8bf;
typedef __attribute__((ext_vector_type(8)))  float  v8f;
typedef __attribute__((ext_vector_type(4)))  unsigned int v4u;
typedef __attribute__((ext_vector_type(8)))  int v8i;
typedef __attribute__((ext_vector_type(4)))  int v4i;

constexpr int Bb = 2, Ss = 2048, Dd = 1024, Hh = 16, HD = 64;
constexpr int RELSPAN = 2 * Ss - 1;   // 4095 relative positions

__device__ inline v16bf join8(v8bf a, v8bf b) {
  return __builtin_shufflevector(a, b, 0,1,2,3,4,5,6,7,8,9,10,11,12,13,14,15);
}

// A-fragment (16x32 bf16): lane<16 holds K{0..7,16..23}, lane>=16 K{8..15,24..31}
__device__ inline v16bf ld_afrag(const __bf16* p) {           // p = row + hb*8
  return join8(*(const v8bf*)p, *(const v8bf*)(p + 16));
}
// B-fragment (32x16 bf16): lane = column, lane<16 holds K{0..7,16..23}, lane>=16 K{8..15,24..31}
__device__ inline v16bf ld_bfrag(const __bf16* p) {           // p = col + hb*8
  return join8(*(const v8bf*)p, *(const v8bf*)(p + 16));
}

// ------------------------------------------------------------- TDM 2D load
// Build a D# per CDNA5 ISA §8 and issue TENSOR_LOAD_TO_LDS (6-arg form).
__device__ inline void tdm_load_2d(unsigned lds_addr, const void* gaddr,
                                   unsigned tensor_d0, unsigned tensor_d1,
                                   unsigned tile_d0, unsigned tile_d1,
                                   unsigned stride0_elems) {
  unsigned long long ga = (unsigned long long)(uintptr_t)gaddr;
  v4u g0;
  g0[0] = 1u;                                    // count=1 (valid), user mode
  g0[1] = lds_addr;                              // LDS byte address
  g0[2] = (unsigned)(ga & 0xffffffffu);          // global_addr[31:0]
  g0[3] = (unsigned)((ga >> 32) & 0x01ffffffu)   // global_addr[56:32]
          | (2u << 30);                          // type = 2 ("image")
  v8i g1;
  g1[0] = (int)(1u << 16);                       // data_size=1 -> 2-byte elems
  g1[1] = (int)((tensor_d0 & 0xffffu) << 16);    // tensor_dim0[15:0] @ bits63:48
  g1[2] = (int)(((tensor_d0 >> 16) & 0xffffu) |  // tensor_dim0[31:16]
                ((tensor_d1 & 0xffffu) << 16));  // tensor_dim1[15:0]
  g1[3] = (int)(((tensor_d1 >> 16) & 0xffffu) |  // tensor_dim1[31:16]
                ((tile_d0 & 0xffffu) << 16));    // tile_dim0
  g1[4] = (int)(tile_d1 & 0xffffu);              // tile_dim1 (tile_dim2 = 0)
  g1[5] = (int)stride0_elems;                    // tensor_dim0_stride[31:0]
  g1[6] = 0;                                     // stride0 hi / dim1_stride lo
  g1[7] = 0;
  v4i z4 = {};
  v8i z8 = {};
  __builtin_amdgcn_tensor_load_to_lds(g0, g1, z4, z4, z8, 0);
}

// ---------------------------------------------------------------- convert
__global__ void cvt_f32_bf16(const float* __restrict__ in,
                             __bf16* __restrict__ out, int n) {
  int i = blockIdx.x * blockDim.x + threadIdx.x;
  int stride = gridDim.x * blockDim.x;
  for (; i < n; i += stride) out[i] = (__bf16)in[i];
}

// ------------------------------------------------- T5 rel-bias LUT [H,4095]
__global__ void build_bias(const float* __restrict__ tbl,
                           float* __restrict__ biasA) {
  int idx = blockIdx.x * blockDim.x + threadIdx.x;
  if (idx >= Hh * RELSPAN) return;
  int h = idx / RELSPAN;
  int rel = idx % RELSPAN - (Ss - 1);   // rel = k - q
  int n = -rel;
  int ret = (n < 0) ? 16 : 0;
  int an = n < 0 ? -n : n;
  int bucket;
  if (an < 8) bucket = ret + an;
  else {
    int v = 8 + (int)(logf((float)an * 0.125f) * (8.0f / logf(16.0f)));
    bucket = ret + (v > 15 ? 15 : v);
  }
  biasA[idx] = tbl[bucket * Hh + h];
}

// ---------------------------------------------------------------- QKV GEMM
// 32x64 output tile per wave (8 accumulators), NT, K-contiguous A and W.
__global__ void __launch_bounds__(256) gemm_qkv(
    const __bf16* __restrict__ A,   // [M=B*S, K=D]
    const __bf16* __restrict__ Wt,  // [N=D,  K=D]
    const float*  __restrict__ bias,
    __bf16* __restrict__ out, int transpose_v) {
  const int K = Dd;
  int lane = threadIdx.x & 31, ln = lane & 15, hb = lane >> 4;
  int flat = blockIdx.x * 8 + (threadIdx.x >> 5);
  const int tn = Dd / 64;                  // 16 N-groups
  int m0 = (flat / tn) * 32;
  int n0 = (flat % tn) * 64;

  const __bf16* arow0 = A + (size_t)(m0 + ln) * K + hb * 8;
  const __bf16* arow1 = A + (size_t)(m0 + 16 + ln) * K + hb * 8;
  const __bf16* brow  = Wt + (size_t)(n0 + ln) * K + hb * 8;
  v8f acc[2][4] = {};
  for (int kk = 0; kk < K; kk += 32) {
    v16bf af0 = ld_afrag(arow0 + kk);
    v16bf af1 = ld_afrag(arow1 + kk);
#pragma unroll
    for (int j = 0; j < 4; j++) {
      v16bf bf = ld_bfrag(brow + (size_t)j * 16 * K + kk);
      acc[0][j] = __builtin_amdgcn_wmma_f32_16x16x32_bf16(false, af0, false, bf,
                                                          (short)0, acc[0][j],
                                                          false, false);
      acc[1][j] = __builtin_amdgcn_wmma_f32_16x16x32_bf16(false, af1, false, bf,
                                                          (short)0, acc[1][j],
                                                          false, false);
    }
  }
#pragma unroll
  for (int mi = 0; mi < 2; mi++)
#pragma unroll
    for (int j = 0; j < 4; j++) {
      int col = n0 + j * 16 + ln;
      float bv = bias[col];
      int hcol = col >> 6, hd = col & 63;
#pragma unroll
      for (int i = 0; i < 8; i++) {
        int row = m0 + mi * 16 + i + 8 * hb;
        int bidx = row >> 11, s = row & (Ss - 1);
        float val = acc[mi][j][i] + bv;
        size_t idx;
        if (!transpose_v)
          idx = (((size_t)bidx * Hh + hcol) * Ss + s) * HD + hd;
        else
          idx = (((size_t)bidx * Hh + hcol) * HD + hd) * (size_t)Ss + s;
        out[idx] = (__bf16)val;
      }
    }
}

// ---------------------------------------------------------------- attention
// Block = one (b,h) and 128 q rows (8 waves x 16). K/V^T tiles stream
// through double-buffered LDS via the TDM: wave 0 issues tile kt+1 while
// all waves consume tile kt (TDM ops complete in order per wave, so
// tensorcnt<=2 guarantees the previous pair has landed).
__global__ void __launch_bounds__(256) attn_kernel(
    const __bf16* __restrict__ Q,    // [B*H, S, HD]
    const __bf16* __restrict__ Km,   // [B*H, S, HD]
    const __bf16* __restrict__ Vt,   // [B*H, HD, S]
    const unsigned char* __restrict__ mask,  // [B, S] bool
    const float* __restrict__ biasA,         // [H, 4095]
    __bf16* __restrict__ ctx) {              // [B*H, S, HD]
  __shared__ __bf16 ldsK[2][32 * HD];    // 2 x 4 KB: 32 keys x 64 hd
  __shared__ __bf16 ldsV[2][HD * 32];    // 2 x 4 KB: 64 hd rows x 32 keys
  __shared__ __bf16 ldsP[8 * 16 * 32];   // 8 KB: per-wave P tiles

  int lane = threadIdx.x & 31, ln = lane & 15, hb = lane >> 4;
  int wid = threadIdx.x >> 5;
  int bh = blockIdx.x >> 4;              // 16 q-blocks per (b,h)
  int qblk = blockIdx.x & 15;
  int h = bh & (Hh - 1);
  int b = bh >> 4;
  int q0 = qblk * 128 + wid * 16;

  const __bf16* Qh = Q  + (size_t)bh * Ss * HD;
  const __bf16* Kh = Km + (size_t)bh * Ss * HD;
  const __bf16* Vh = Vt + (size_t)bh * HD * Ss;
  const float*  bh_bias = biasA + h * RELSPAN + (Ss - 1);
  __bf16* lp = ldsP + wid * 16 * 32;
  const int NT = Ss / 32;                // 64 kv tiles

  v16bf aq[2];
#pragma unroll
  for (int t = 0; t < 2; t++)
    aq[t] = ld_afrag(Qh + (size_t)(q0 + ln) * HD + t * 32 + hb * 8);

  v8f acc[4] = {};
  float mrow[8], lrow[8];
#pragma unroll
  for (int i = 0; i < 8; i++) { mrow[i] = -1e30f; lrow[i] = 0.f; }

  // preload tile 0 into buffer 0
  if (wid == 0) {
    tdm_load_2d((unsigned)(uintptr_t)ldsK[0], Kh, HD, 32, HD, 32, HD);
    tdm_load_2d((unsigned)(uintptr_t)ldsV[0], Vh, 32, HD, 32, HD, Ss);
  }

  for (int kt = 0; kt < NT; kt++) {
    int k0 = kt * 32;
    int buf = kt & 1;
    if (wid == 0) {
      if (kt + 1 < NT) {
        // stream next tile into the other buffer while this one is consumed
        tdm_load_2d((unsigned)(uintptr_t)ldsK[buf ^ 1],
                    Kh + (size_t)(k0 + 32) * HD, HD, 32, HD, 32, HD);
        tdm_load_2d((unsigned)(uintptr_t)ldsV[buf ^ 1],
                    Vh + (k0 + 32), 32, HD, 32, HD, Ss);
        __builtin_amdgcn_s_wait_tensorcnt(2);   // current tile's pair done
      } else {
        __builtin_amdgcn_s_wait_tensorcnt(0);
      }
    } else if (wid == 1 && k0 + 64 < Ss) {
      __builtin_prefetch(Kh + (size_t)(k0 + 64) * HD + lane * 64, 0, 1);
      __builtin_prefetch(Vh + (size_t)lane * Ss + k0 + 64, 0, 1);
      __builtin_prefetch(Vh + (size_t)(lane + 32) * Ss + k0 + 64, 0, 1);
    }
    __syncthreads();   // tile `buf` visible to all waves

    // scores: two 16x16 tiles over K=HD=64
    v8f st[2];
#pragma unroll
    for (int t = 0; t < 2; t++) {
      v8f s = {};
#pragma unroll
      for (int ks = 0; ks < 2; ks++) {
        v16bf bf = ld_bfrag(ldsK[buf] + (t * 16 + ln) * HD + ks * 32 + hb * 8);
        s = __builtin_amdgcn_wmma_f32_16x16x32_bf16(false, aq[ks], false, bf,
                                                    (short)0, s, false, false);
      }
      st[t] = s;
    }

    // relative-position bias (LUT) + mask
#pragma unroll
    for (int t = 0; t < 2; t++) {
      int kpos = k0 + t * 16 + ln;
      unsigned char mk = mask[b * Ss + kpos];
#pragma unroll
      for (int i = 0; i < 8; i++) {
        int qpos = q0 + i + 8 * hb;
        float sc = st[t][i] + bh_bias[kpos - qpos];
        st[t][i] = mk ? -3.0e38f : sc;
      }
    }

    // online softmax (row spread across 16 lanes)
#pragma unroll
    for (int i = 0; i < 8; i++) {
      float rm = fmaxf(st[0][i], st[1][i]);
#pragma unroll
      for (int off = 1; off < 16; off <<= 1)
        rm = fmaxf(rm, __shfl_xor(rm, off, 32));
      float newm = fmaxf(mrow[i], rm);
      float scale = __expf(mrow[i] - newm);
      float p0 = __expf(st[0][i] - newm);
      float p1 = __expf(st[1][i] - newm);
      float rs = p0 + p1;
#pragma unroll
      for (int off = 1; off < 16; off <<= 1)
        rs += __shfl_xor(rs, off, 32);
      mrow[i] = newm;
      lrow[i] = lrow[i] * scale + rs;
      st[0][i] = p0; st[1][i] = p1;
#pragma unroll
      for (int c = 0; c < 4; c++) acc[c][i] *= scale;
    }

    // P (C-layout) -> wave-private LDS 16x32 -> reload as A-fragment.
    // LDS ops from one wave execute in order (DScnt), so only a compiler
    // scheduling barrier is needed (no block barrier, no wait).
#pragma unroll
    for (int t = 0; t < 2; t++)
#pragma unroll
      for (int i = 0; i < 8; i++)
        lp[(i + 8 * hb) * 32 + t * 16 + ln] = (__bf16)st[t][i];
    __builtin_amdgcn_wave_barrier();
    v16bf pf = ld_afrag(lp + ln * 32 + hb * 8);

#pragma unroll
    for (int c = 0; c < 4; c++) {
      v16bf vf = ld_bfrag(ldsV[buf] + (c * 16 + ln) * 32 + hb * 8);
      acc[c] = __builtin_amdgcn_wmma_f32_16x16x32_bf16(false, pf, false, vf,
                                                       (short)0, acc[c],
                                                       false, false);
    }
    __syncthreads();   // all reads of `buf` done before it is refilled
  }

#pragma unroll
  for (int c = 0; c < 4; c++)
#pragma unroll
    for (int i = 0; i < 8; i++) {
      int row = q0 + i + 8 * hb;
      float val = acc[c][i] / lrow[i];
      ctx[(size_t)bh * Ss * HD + (size_t)row * HD + c * 16 + ln] = (__bf16)val;
    }
}

// ---------------------------------------------------------------- out proj
__global__ void __launch_bounds__(256) gemm_out(
    const __bf16* __restrict__ Ctx,  // [B*H, S, HD]
    const __bf16* __restrict__ Wt,   // [D, D]
    const float*  __restrict__ bias,
    float* __restrict__ out) {
  const int K = Dd;
  int lane = threadIdx.x & 31, ln = lane & 15, hb = lane >> 4;
  int flat = blockIdx.x * 8 + (threadIdx.x >> 5);
  const int tn = Dd / 64;
  int m0 = (flat / tn) * 32;
  int n0 = (flat % tn) * 64;
  int row0 = m0 + ln;
  int b = row0 >> 11;
  int s0 = row0 & (Ss - 1), s1 = s0 + 16;
  const __bf16* brow = Wt + (size_t)(n0 + ln) * K + hb * 8;
  v8f acc[2][4] = {};
  for (int kk = 0; kk < K; kk += 32) {
    int kb0 = kk + hb * 8, kb1 = kb0 + 16;   // 8-chunks stay within one head
    const __bf16* h0 = Ctx + ((size_t)(b * Hh + (kb0 >> 6)) * Ss) * HD + (kb0 & 63);
    const __bf16* h1 = Ctx + ((size_t)(b * Hh + (kb1 >> 6)) * Ss) * HD + (kb1 & 63);
    v16bf af0 = join8(*(const v8bf*)(h0 + (size_t)s0 * HD),
                      *(const v8bf*)(h1 + (size_t)s0 * HD));
    v16bf af1 = join8(*(const v8bf*)(h0 + (size_t)s1 * HD),
                      *(const v8bf*)(h1 + (size_t)s1 * HD));
#pragma unroll
    for (int j = 0; j < 4; j++) {
      v16bf bf = ld_bfrag(brow + (size_t)j * 16 * K + kk);
      acc[0][j] = __builtin_amdgcn_wmma_f32_16x16x32_bf16(false, af0, false, bf,
                                                          (short)0, acc[0][j],
                                                          false, false);
      acc[1][j] = __builtin_amdgcn_wmma_f32_16x16x32_bf16(false, af1, false, bf,
                                                          (short)0, acc[1][j],
                                                          false, false);
    }
  }
#pragma unroll
  for (int mi = 0; mi < 2; mi++)
#pragma unroll
    for (int j = 0; j < 4; j++) {
      float bv = bias[n0 + j * 16 + ln];
#pragma unroll
      for (int i = 0; i < 8; i++) {
        int r = m0 + mi * 16 + i + 8 * hb;
        out[(size_t)r * Dd + n0 + j * 16 + ln] = acc[mi][j][i] + bv;
      }
    }
}

// ---------------------------------------------------------------- launch
extern "C" void kernel_launch(void* const* d_in, const int* in_sizes, int n_in,
                              void* d_out, int out_size, void* d_ws, size_t ws_size,
                              hipStream_t stream) {
  const float* X  = (const float*)d_in[0];
  const unsigned char* mask = (const unsigned char*)d_in[1];
  const float* Wq = (const float*)d_in[2];
  const float* bq = (const float*)d_in[3];
  const float* Wk = (const float*)d_in[4];
  const float* bk = (const float*)d_in[5];
  const float* Wv = (const float*)d_in[6];
  const float* bv = (const float*)d_in[7];
  const float* Wo = (const float*)d_in[8];
  const float* bo = (const float*)d_in[9];
  const float* tbl = (const float*)d_in[10];

  char* ws = (char*)d_ws;
  const size_t MB = (size_t)1 << 20;
  __bf16* Xb  = (__bf16*)(ws + 0);        //  8 MB  [B*S, D]
  __bf16* Wqb = (__bf16*)(ws + 8  * MB);
  __bf16* Wkb = (__bf16*)(ws + 10 * MB);
  __bf16* Wvb = (__bf16*)(ws + 12 * MB);
  __bf16* Wob = (__bf16*)(ws + 14 * MB);
  __bf16* Qb  = (__bf16*)(ws + 16 * MB);  //  8 MB  [B*H, S, HD]
  __bf16* Kb  = (__bf16*)(ws + 24 * MB);  //  8 MB  [B*H, S, HD]
  __bf16* Vtb = (__bf16*)(ws + 32 * MB);  //  8 MB  [B*H, HD, S]
  __bf16* Ctx = (__bf16*)(ws + 40 * MB);  //  8 MB  [B*H, S, HD]
  float*  BiasA = (float*)(ws + 48 * MB); //  256 KB  [H, 4095]

  cvt_f32_bf16<<<2048, 256, 0, stream>>>(X,  Xb,  Bb * Ss * Dd);
  cvt_f32_bf16<<<1024, 256, 0, stream>>>(Wq, Wqb, Dd * Dd);
  cvt_f32_bf16<<<1024, 256, 0, stream>>>(Wk, Wkb, Dd * Dd);
  cvt_f32_bf16<<<1024, 256, 0, stream>>>(Wv, Wvb, Dd * Dd);
  cvt_f32_bf16<<<1024, 256, 0, stream>>>(Wo, Wob, Dd * Dd);
  build_bias<<<(Hh * RELSPAN + 255) / 256, 256, 0, stream>>>(tbl, BiasA);

  const int wtiles = (Bb * Ss / 32) * (Dd / 64);  // 2048 wave tiles
  gemm_qkv<<<wtiles / 8, 256, 0, stream>>>(Xb, Wqb, bq, Qb, 0);
  gemm_qkv<<<wtiles / 8, 256, 0, stream>>>(Xb, Wkb, bk, Kb, 0);
  gemm_qkv<<<wtiles / 8, 256, 0, stream>>>(Xb, Wvb, bv, Vtb, 1);

  attn_kernel<<<Bb * Hh * (Ss / 128), 256, 0, stream>>>(Qb, Kb, Vtb, mask,
                                                        BiasA, Ctx);

  gemm_out<<<wtiles / 8, 256, 0, stream>>>(Ctx, Wob, bo, (float*)d_out);
}
